// KMeans_19370302505014
// MI455X (gfx1250) — matrix operand order repaired and checked by reading, not compiled, not run on hardware
//
#include <hip/hip_runtime.h>
#include <math.h>

// ---------------------------------------------------------------------------
// KMeans summarize for MI455X (gfx1250, wave32, WMMA)
//   distances GEMM via V_WMMA_F32_16X16X32_BF16 with fp32->bf16 hi/lo split
//   (3 WMMAs per tile-k-step ~= fp32 accuracy at bf16 matrix throughput)
//   argmin via packed u64 ds_min LDS atomics, segment-sum via ds_add_f32
//   into LDS-resident (K x D) tables (256KB of the 320KB WGP LDS), flushed
//   once per block with global fp32 atomics.
// ---------------------------------------------------------------------------

typedef __attribute__((ext_vector_type(16))) __bf16 v16bf;
typedef __attribute__((ext_vector_type(8)))  float  v8f;

#define NROWS   500000
#define DDIM    64
#define KCENT   512
#define NTILES  (NROWS / 16)     // 31250, exact
#define GRID_BLOCKS 256

__device__ __forceinline__ v8f wmma_bf16(v16bf a, v16bf b, v8f c) {
  // (neg_a, A, neg_b, B, c_mod, C, reuse_a, reuse_b)
  return __builtin_amdgcn_wmma_f32_16x16x32_bf16(false, a, false, b,
                                                 (short)0, c, false, false);
}

__device__ __forceinline__ void split_bf16(float v, __bf16& hi, __bf16& lo) {
  hi = (__bf16)v;
  lo = (__bf16)(v - (float)hi);
}

__global__ void __launch_bounds__(256) zero_out_kernel(float* out, int n) {
  int i = blockIdx.x * blockDim.x + threadIdx.x;
  if (i < n) out[i] = 0.0f;
}

__global__ void __launch_bounds__(256, 1)
kmeans_kernel(const float* __restrict__ X,
              const float* __restrict__ C,
              const float* __restrict__ SW,
              float* __restrict__ out) {
  // ---- LDS: 2 accumulator tables (256KB) + small staging ----
  __shared__ float s_xw[KCENT * DDIM];      // 128 KB
  __shared__ float s_w [KCENT * DDIM];      // 128 KB
  __shared__ float s_cc[KCENT];             // centroid squared norms
  __shared__ unsigned long long s_min[16];  // packed (key<<32 | centroid)
  __shared__ int   s_y[16];
  __shared__ float s_inertia[16];

  const int tid  = threadIdx.x;        // 0..255
  const int wave = tid >> 5;           // 0..7
  const int lane = tid & 31;
  const int g    = lane >> 4;          // half-wave group (WMMA layout)
  const int nn   = lane & 15;

  // ---- zero LDS accumulators ----
  for (int i = tid; i < KCENT * DDIM; i += 256) { s_xw[i] = 0.0f; s_w[i] = 0.0f; }

  // ---- centroid squared norms ----
  for (int c = tid; c < KCENT; c += 256) {
    const float* cp = C + c * DDIM;
    float s = 0.0f;
#pragma unroll
    for (int j = 0; j < DDIM; ++j) s = fmaf(cp[j], cp[j], s);
    s_cc[c] = s;
  }
  __syncthreads();

  // ---- register-resident B fragments: each wave owns 4 tiles of 16 centroids
  //      B layout (32x16 bf16): lane (g,nn): elems 0..7 <-> K = 32*ks + 8*g + e ; elems 8..15 <-> K + 16
  v16bf Bhi[4][2], Blo[4][2];
  float ccv[4];
#pragma unroll
  for (int tc = 0; tc < 4; ++tc) {
    const int cidx = ((wave << 2) + tc) * 16 + nn;
    ccv[tc] = s_cc[cidx];
    const float* cp = C + cidx * DDIM;
#pragma unroll
    for (int ks = 0; ks < 2; ++ks) {
#pragma unroll
      for (int e = 0; e < 16; ++e) {
        __bf16 h, l;
        split_bf16(cp[ks * 32 + g * 8 + (e & 7) + (e >> 3) * 16], h, l);
        Bhi[tc][ks][e] = h;
        Blo[tc][ks][e] = l;
      }
    }
  }

  float my_inertia = 0.0f;   // live only in tid<16

  for (int t = blockIdx.x; t < NTILES; t += gridDim.x) {
    const int row0 = t * 16;

    if (tid < 16) s_min[tid] = ~0ull;
    __syncthreads();

    // ---- A fragments (16x32 bf16): lane (g,nn) holds row nn,
    //      elems 0..7 <-> K = 32*ks + 8*g + e ; elems 8..15 <-> K + 16
    v16bf Ahi[2], Alo[2];
    const float* xrow = X + (size_t)(row0 + nn) * DDIM;
#pragma unroll
    for (int ks = 0; ks < 2; ++ks) {
#pragma unroll
      for (int e = 0; e < 8; ++e) {
        __bf16 h, l;
        split_bf16(xrow[ks * 32 + g * 8 + e], h, l);
        Ahi[ks][e] = h; Alo[ks][e] = l;
        split_bf16(xrow[ks * 32 + 16 + g * 8 + e], h, l);
        Ahi[ks][e + 8] = h; Alo[ks][e + 8] = l;
      }
    }

    // ---- GEMM + per-lane running min of (||c||^2 - 2 x.c) ----
    float bestv[8]; int bestn[8];
#pragma unroll
    for (int i = 0; i < 8; ++i) { bestv[i] = 3.0e38f; bestn[i] = 0; }

#pragma unroll
    for (int tc = 0; tc < 4; ++tc) {
      v8f acc = (v8f)(0.0f);
#pragma unroll
      for (int ks = 0; ks < 2; ++ks) {
        acc = wmma_bf16(Ahi[ks], Bhi[tc][ks], acc);   // hi*hi
        acc = wmma_bf16(Ahi[ks], Blo[tc][ks], acc);   // hi*lo
        acc = wmma_bf16(Alo[ks], Bhi[tc][ks], acc);   // lo*hi
      }
      const float cc   = ccv[tc];
      const int   nidx = ((wave << 2) + tc) * 16 + nn;
#pragma unroll
      for (int i = 0; i < 8; ++i) {
        float v = fmaf(-2.0f, acc[i], cc);            // row i + 8*g, col nidx
        if (v < bestv[i]) { bestv[i] = v; bestn[i] = nidx; }
      }
    }

    // ---- pack (monotonic float key, centroid idx) and ds_min_u64 reduce ----
#pragma unroll
    for (int i = 0; i < 8; ++i) {
      unsigned int key = __float_as_uint(bestv[i]);
      key ^= ((unsigned int)(-(int)(key >> 31)) | 0x80000000u);
      unsigned long long p =
          ((unsigned long long)key << 32) | (unsigned int)bestn[i];
      atomicMin(&s_min[i + (g << 3)], p);
    }
    __syncthreads();

    // ---- per-row finalize: ||x||^2, min distance, inertia, assignment ----
    if (tid < 16) {
      const float* xr = X + (size_t)(row0 + tid) * DDIM;
      float xx = 0.0f;
#pragma unroll
      for (int j = 0; j < DDIM; ++j) xx = fmaf(xr[j], xr[j], xx);
      unsigned long long p = s_min[tid];
      unsigned int k2 = (unsigned int)(p >> 32);
      unsigned int key = (k2 & 0x80000000u) ? (k2 ^ 0x80000000u) : ~k2;
      float v  = __uint_as_float(key);
      float d2 = fmaxf(xx + v, 0.0f) * (1.0f / (float)DDIM);
      my_inertia += sqrtf(d2);
      s_y[tid] = (int)(p & 0xFFFFFFFFu);
    }
    __syncthreads();

    // ---- scatter: 16 rows x 64 cols, ds_add_f32 into LDS tables ----
#pragma unroll
    for (int e = 0; e < 4; ++e) {
      const int idx = e * 256 + tid;        // 0..1023, lane-contiguous
      const int r   = idx >> 6;
      const int col = idx & 63;
      const int y   = s_y[r];
      const size_t goff = (size_t)(row0 + r) * DDIM + col;
      const float w = SW[goff];
      const float x = X[goff];
      atomicAdd(&s_w [y * DDIM + col], w);
      atomicAdd(&s_xw[y * DDIM + col], x * w);
    }
    __syncthreads();
  }

  // ---- flush block-local tables + inertia to global output ----
  if (tid < 16) s_inertia[tid] = my_inertia;
  __syncthreads();

  for (int i = tid; i < KCENT * DDIM; i += 256) {
    atomicAdd(&out[i],                 s_xw[i]);   // rows [0, K)   : xw_sum
    atomicAdd(&out[KCENT * DDIM + i],  s_w[i]);    // rows [K, 2K)  : w_sum
  }

  float bi = 0.0f;
#pragma unroll
  for (int r = 0; r < 16; ++r) bi += s_inertia[r];
  if (tid < DDIM) atomicAdd(&out[2 * KCENT * DDIM + tid], bi);  // inertia row
}

extern "C" void kernel_launch(void* const* d_in, const int* in_sizes, int n_in,
                              void* d_out, int out_size, void* d_ws, size_t ws_size,
                              hipStream_t stream) {
  (void)in_sizes; (void)n_in; (void)d_ws; (void)ws_size;
  const float* X  = (const float*)d_in[0];
  const float* C  = (const float*)d_in[1];
  const float* SW = (const float*)d_in[2];
  float* out = (float*)d_out;

  // output is accumulated atomically -> must be zeroed every launch
  zero_out_kernel<<<(out_size + 255) / 256, 256, 0, stream>>>(out, out_size);
  kmeans_kernel<<<GRID_BLOCKS, 256, 0, stream>>>(X, C, SW, out);
}
